// LucidVectorQuantizer_48833778155970
// MI455X (gfx1250) — matrix operand order repaired and checked by reading, not compiled, not run on hardware
//
#include <hip/hip_runtime.h>
#include <stdint.h>

// Problem constants (from reference)
#define BB   16
#define NN   4096
#define DD   256
#define KK   2048
#define TT   (BB * NN)          // 65536 tokens
#define COMMIT_SCALE 0.25f

// ---- WMMA vector types (gfx1250, wave32) ----
typedef __attribute__((ext_vector_type(16))) __bf16       v16bf;
typedef __attribute__((ext_vector_type(8)))  float        v8f;
typedef __attribute__((ext_vector_type(4)))  unsigned int v4u;
typedef __attribute__((ext_vector_type(4)))  int          v4i;

union FragBF {
    v4u   q[2];   // 2 x 16B LDS loads
    v16bf v;      // 16 bf16 per lane = one 16x32 (A) or 32x16 (B) operand
};

// ---- LDS layout (bytes). Rows padded to 264 bf16 (528B = 33*16). ----
#define ROWB      528
#define A_HI_OFF  0
#define A_LO_OFF  (128 * ROWB)               //  67584
#define BHALF     (64 * ROWB)                //  33792 (one matrix of a buffer)
#define BUFB      (2 * BHALF)                //  67584 (hi+lo)
#define B_BASE    (2 * 128 * ROWB)           // 135168
#define E2L_OFF   (B_BASE + 2 * BUFB)        // 270336
#define SMEM_BYTES (E2L_OFF + KK * 4)        // 278528  (< 320KB WGP LDS)

// ---- CDNA5 async global->LDS path (guarded; sync-copy fallback) ----
#if defined(__has_builtin)
#if __has_builtin(__builtin_amdgcn_global_load_async_to_lds_b128)
#define HAVE_ASYNC_LDS 1
#endif
#endif
#ifndef HAVE_ASYNC_LDS
#define HAVE_ASYNC_LDS 0
#endif

// Builtin signature (from hipcc diagnostic): arg0 is AS1 (global) pointer to
// 4xi32 vector; arg1 is the LDS-side pointer (AS3), then imm offset, cpol.
typedef __attribute__((address_space(1))) v4i* g_v4i_p;   // global int4*
typedef __attribute__((address_space(3))) v4i* l_v4i_p;   // LDS int4*

__device__ __forceinline__ g_v4i_p as_global_v4(const void* p) {
    return (g_v4i_p)(uintptr_t)p;
}
__device__ __forceinline__ l_v4i_p as_local_v4(void* p) {
    // generic LDS pointer low 32 bits are the LDS byte offset
    return (l_v4i_p)(uint32_t)(uintptr_t)p;
}

__device__ __forceinline__ void wait_async0() {
#if HAVE_ASYNC_LDS
#if __has_builtin(__builtin_amdgcn_s_wait_asynccnt)
    __builtin_amdgcn_s_wait_asynccnt(0);
#else
    asm volatile("s_wait_asynccnt 0" ::: "memory");
#endif
#endif
}

// Split fp32 into bf16 hi (truncated) + bf16 lo (residual).
__device__ __forceinline__ void split_bf16(float x, unsigned short& h, unsigned short& l) {
    unsigned int u  = __float_as_uint(x);
    unsigned int hu = u & 0xFFFF0000u;
    float        rf = x - __uint_as_float(hu);
    h = (unsigned short)(hu >> 16);
    l = (unsigned short)(__float_as_uint(rf) >> 16);
}

// ---------- prep: embed [D,K] -> eT f32 [K,D], ehT/elT bf16 [K,D] ----------
__global__ __launch_bounds__(256) void vq_embed_prep_kernel(
    const float* __restrict__ e, float* __restrict__ eT,
    unsigned short* __restrict__ ehT, unsigned short* __restrict__ elT) {
    int i = blockIdx.x * 256 + threadIdx.x;          // 512 blocks
    for (int t = i; t < DD * KK; t += 512 * 256) {
        int d = t >> 11;            // / KK
        int k = t & (KK - 1);
        float v = e[t];
        unsigned short h, l;
        split_bf16(v, h, l);
        size_t o = (size_t)k * DD + d;
        eT[o]  = v;
        ehT[o] = h;
        elT[o] = l;
    }
}

// ---------- prep: e2[k] = sum_d embed[d][k]^2 ----------
__global__ __launch_bounds__(256) void vq_e2_kernel(
    const float* __restrict__ e, float* __restrict__ e2g) {
    int k = blockIdx.x * 256 + threadIdx.x;          // KK/256 blocks
    float s = 0.f;
    for (int d = 0; d < DD; ++d) {
        float v = e[(size_t)d * KK + k];
        s = fmaf(v, v, s);
    }
    e2g[k] = s;
}

// Stage one 64-code B tile (pre-split bf16 hi/lo) into LDS buffer p.
// Pure 16B-chunk copy: async global->LDS when available.
__device__ __forceinline__ void stage_b_tile(
    char* smem, const unsigned short* __restrict__ ehT,
    const unsigned short* __restrict__ elT, int c0, int p, int tid) {
    const unsigned base = B_BASE + (unsigned)p * BUFB;
#pragma unroll
    for (int j = 0; j < 8; ++j) {
        int c   = j * 256 + tid;            // 2048 chunks of 16B per matrix
        int row = c >> 5;                   // code row 0..63
        int off = (c & 31) * 16;            // byte offset within 512B row
        size_t gOff = (size_t)(c0 + row) * (DD * 2) + off;
        char* lHi = smem + base + row * ROWB + off;
        char* lLo = lHi + BHALF;
#if HAVE_ASYNC_LDS
        __builtin_amdgcn_global_load_async_to_lds_b128(
            as_global_v4((const char*)ehT + gOff), as_local_v4(lHi), 0, 0);
        __builtin_amdgcn_global_load_async_to_lds_b128(
            as_global_v4((const char*)elT + gOff), as_local_v4(lLo), 0, 0);
#else
        *(v4u*)lHi = *(const v4u*)((const char*)ehT + gOff);
        *(v4u*)lLo = *(const v4u*)((const char*)elT + gOff);
#endif
    }
}

// ---------- main: distances via bf16x3 WMMA + running argmin ----------
__global__ __launch_bounds__(256) void vq_argmin_kernel(
    const float* __restrict__ x, const unsigned short* __restrict__ ehT,
    const unsigned short* __restrict__ elT, const float* __restrict__ e2g,
    int* __restrict__ idxInt, float* __restrict__ idxOutF) {
    extern __shared__ char smem[];
    const int tid = threadIdx.x;
    const int t0  = blockIdx.x * 128;                // 128 tokens per block

    // ---- Stage A: x tile [128 x 256] -> bf16 hi/lo in LDS (once) ----
    for (int row = 0; row < 128; ++row) {
        float v = x[(size_t)(t0 + row) * DD + tid];  // coalesced
        unsigned short h, l;
        split_bf16(v, h, l);
        *(unsigned short*)(smem + A_HI_OFF + row * ROWB + tid * 2) = h;
        *(unsigned short*)(smem + A_LO_OFF + row * ROWB + tid * 2) = l;
    }
    // ---- e2 table (all 2048 codes) into LDS once ----
    for (int j = tid; j < KK; j += 256)
        ((float*)(smem + E2L_OFF))[j] = e2g[j];

    // ---- kick off async staging of B tile 0 into buffer 0 ----
    stage_b_tile(smem, ehT, elT, 0, 0, tid);

    const int lane     = tid & 31;
    const int wid      = tid >> 5;                   // 8 waves: rows wid*16..+15
    const int laneHalf = lane >> 4;                  // 0|1
    const int l15      = lane & 15;
    const int rowA     = wid * 16 + l15;             // A-operand row for this lane
    const unsigned aHiBase = A_HI_OFF + rowA * ROWB;
    const unsigned aLoBase = A_LO_OFF + rowA * ROWB;

    wait_async0();
    __syncthreads();   // A, e2 and B tile 0 visible to all waves

    // ---- hoist A fragments (full D) into registers: 16 frags ----
    FragBF ahr[8], alr[8];
#pragma unroll
    for (int ks = 0; ks < 8; ++ks) {
        // A 16x32 bf16 layout: lane<16 -> K {0..7,16..23}; lane>=16 -> +8
        const unsigned ka = (unsigned)(ks * 32 + laneHalf * 8) * 2u;
        ahr[ks].q[0] = *(const v4u*)(smem + aHiBase + ka);
        ahr[ks].q[1] = *(const v4u*)(smem + aHiBase + ka + 32);
        alr[ks].q[0] = *(const v4u*)(smem + aLoBase + ka);
        alr[ks].q[1] = *(const v4u*)(smem + aLoBase + ka + 32);
    }

    float bestd[8];
    int   bestc[8];
#pragma unroll
    for (int r = 0; r < 8; ++r) { bestd[r] = 3.4e38f; bestc[r] = 0; }

    const float* e2s = (const float*)(smem + E2L_OFF);

    for (int tile = 0; tile < 32; ++tile) {
        const int c0 = tile * 64;
        const int p  = tile & 1;

        // overlap: issue async staging of the next tile into the other buffer
        if (tile + 1 < 32)
            stage_b_tile(smem, ehT, elT, c0 + 64, 1 - p, tid);

        // ---- 16x64 tile per wave: 8 k-steps x 4 subtiles x 3 WMMA ----
        const unsigned bb = B_BASE + (unsigned)p * BUFB;
        v8f acc[4] = {};
#pragma unroll
        for (int ks = 0; ks < 8; ++ks) {
            // B 32x16 bf16 layout: lane<16 -> K {0..7,16..23}; lane>=16 -> +8
            const unsigned kb = (unsigned)(ks * 32 + laneHalf * 8) * 2u;
#pragma unroll
            for (int sub = 0; sub < 4; ++sub) {
                const int code = sub * 16 + l15;
                FragBF bh, bl;
                const char* bp = smem + bb + code * ROWB + kb;
                bh.q[0] = *(const v4u*)bp;
                bh.q[1] = *(const v4u*)(bp + 32);
                const char* bq = bp + BHALF;
                bl.q[0] = *(const v4u*)bq;
                bl.q[1] = *(const v4u*)(bq + 32);
                // dot(x,e) ~= xh*eh + xh*el + xl*eh  (f32 accumulate)
                acc[sub] = __builtin_amdgcn_wmma_f32_16x16x32_bf16(
                    false, ahr[ks].v, false, bh.v, (short)0, acc[sub], false, false);
                acc[sub] = __builtin_amdgcn_wmma_f32_16x16x32_bf16(
                    false, ahr[ks].v, false, bl.v, (short)0, acc[sub], false, false);
                acc[sub] = __builtin_amdgcn_wmma_f32_16x16x32_bf16(
                    false, alr[ks].v, false, bh.v, (short)0, acc[sub], false, false);
            }
        }

        // ---- epilogue: score = |e|^2 - 2*dot (x^2 constant -> dropped) ----
#pragma unroll
        for (int sub = 0; sub < 4; ++sub) {
            const float e2v  = e2s[c0 + sub * 16 + l15];
            const int   code = c0 + sub * 16 + l15;
#pragma unroll
            for (int r = 0; r < 8; ++r) {
                float sc = fmaf(-2.f, acc[sub][r], e2v);
                if (sc < bestd[r]) { bestd[r] = sc; bestc[r] = code; }
            }
        }

        // next tile's async staging must be complete in every wave
        wait_async0();
        __syncthreads();
    }

    // ---- argmin across the 16 lanes holding each C row (wave32 shfl) ----
#pragma unroll
    for (int r = 0; r < 8; ++r) {
        float dv = bestd[r];
        int   cv = bestc[r];
#pragma unroll
        for (int m = 1; m <= 8; m <<= 1) {
            float od = __shfl_xor(dv, m, 32);
            int   oc = __shfl_xor(cv, m, 32);
            if (od < dv || (od == dv && oc < cv)) { dv = od; cv = oc; }
        }
        if (l15 == 0) {
            // C layout: lanes 0-15 hold M=r, lanes 16-31 hold M=r+8
            int token = t0 + wid * 16 + laneHalf * 8 + r;
            idxInt[token]  = cv;
            idxOutF[token] = (float)cv;
        }
    }
}

// ---------- gather quantize + fused commit-loss partials ----------
__global__ __launch_bounds__(256) void vq_gather_loss_kernel(
    const float* __restrict__ x, const float* __restrict__ eT,
    const int* __restrict__ idxI, float* __restrict__ q,
    float* __restrict__ part) {
    __shared__ float wsum[8];
    const int tid = threadIdx.x;
    float s = 0.f;
    for (int i = 0; i < 32; ++i) {               // 2048 blocks * 32 tokens
        int    t   = blockIdx.x * 32 + i;
        int    c   = idxI[t];                    // uniform within block iter
        size_t off = (size_t)t * DD + tid;
        float  qv  = eT[(size_t)c * DD + tid];   // coalesced (L2-resident)
        float  xv  = x[off];
        q[off] = qv;
        float df = qv - xv;
        s = fmaf(df, df, s);
    }
    for (int m = 16; m >= 1; m >>= 1) s += __shfl_xor(s, m, 32);
    if ((tid & 31) == 0) wsum[tid >> 5] = s;
    __syncthreads();
    if (tid == 0) {
        float tot = 0.f;
        for (int w = 0; w < 8; ++w) tot += wsum[w];   // fixed order
        part[blockIdx.x] = tot;
    }
}

// ---------- deterministic loss finalize ----------
__global__ __launch_bounds__(256) void vq_finalize_kernel(
    const float* __restrict__ part, float* __restrict__ loss) {
    __shared__ float acc[256];
    const int tid = threadIdx.x;
    float s = 0.f;
    for (int i = tid; i < 2048; i += 256) s += part[i];  // fixed order
    acc[tid] = s;
    __syncthreads();
    if (tid == 0) {
        float tot = 0.f;
        for (int i = 0; i < 256; ++i) tot += acc[i];     // fixed order
        loss[0] = tot * (COMMIT_SCALE / ((float)TT * (float)DD));
    }
}

extern "C" void kernel_launch(void* const* d_in, const int* in_sizes, int n_in,
                              void* d_out, int out_size, void* d_ws, size_t ws_size,
                              hipStream_t stream) {
    const float* x     = (const float*)d_in[0];     // [B,N,D] fp32
    const float* embed = (const float*)d_in[1];     // [D,K]   fp32
    float* out = (float*)d_out;

    // workspace carve-up (bytes)
    char*           ws   = (char*)d_ws;
    float*          eT   = (float*)ws;                          // 2 MB  f32 [K,D]
    unsigned short* ehT  = (unsigned short*)(ws + 2097152);     // 1 MB  bf16 hi [K,D]
    unsigned short* elT  = (unsigned short*)(ws + 3145728);     // 1 MB  bf16 lo [K,D]
    float*          e2g  = (float*)(ws + 4194304);              // 8 KB
    float*          part = (float*)(ws + 4202496);              // 8 KB
    int*            idxI = (int*)(ws + 4210688);                // 256 KB

    // output packing: quantize | commit_loss | indices(as float)
    float* qOut = out;
    float* loss = out + (size_t)TT * DD;
    float* idxF = loss + 1;

    (void)hipFuncSetAttribute((const void*)vq_argmin_kernel,
                              hipFuncAttributeMaxDynamicSharedMemorySize,
                              SMEM_BYTES);

    vq_embed_prep_kernel<<<512, 256, 0, stream>>>(embed, eT, ehT, elT);
    vq_e2_kernel<<<KK / 256, 256, 0, stream>>>(embed, e2g);
    vq_argmin_kernel<<<TT / 128, 256, SMEM_BYTES, stream>>>(x, ehT, elT, e2g, idxI, idxF);
    vq_gather_loss_kernel<<<2048, 256, 0, stream>>>(x, eT, idxI, qOut, part);
    vq_finalize_kernel<<<1, 256, 0, stream>>>(part, loss);
}
